// Head_17446157156410
// MI455X (gfx1250) — matrix operand order repaired and checked by reading, not compiled, not run on hardware
//
#include <hip/hip_runtime.h>
#include <hip/hip_bf16.h>

// Fused causal single-head attention for MI455X (gfx1250, wave32, WMMA).
// x:[B,64,128] f32, Wq/Wk/Wv:[64,128] f32  ->  out:[B,64,64] f32
// One workgroup (4 waves) per batch element; wave w owns rows 16w..16w+15.

typedef __attribute__((ext_vector_type(16))) _Float16 v16h;
typedef __attribute__((ext_vector_type(8)))  float    v8f;

#define CTX 64
#define EMB 128
#define HSZ 64
#define LDP 72   // halfwords per LDS row (64 + 8 pad); row stride 144B = 9*16B -> b128 aligned

__device__ __forceinline__ v8f wmma_f16f32(v16h a, v16h b, v8f c) {
  // D = A(16x32 f16) * B(32x16 f16) + C(16x16 f32)
  return __builtin_amdgcn_wmma_f32_16x16x32_f16(false, a, false, b, (short)0, c, false, false);
}

// XOR-swizzle within groups of 32 lanes (and=0x1f, or=0, xor in [14:10]).
#define SWZ_XOR(v, imm) \
  __builtin_bit_cast(float, __builtin_amdgcn_ds_swizzle(__builtin_bit_cast(int, (v)), (imm)))

__device__ __forceinline__ float red16_max(float v) {
  v = fmaxf(v, SWZ_XOR(v, 0x041f));  // xor 1
  v = fmaxf(v, SWZ_XOR(v, 0x081f));  // xor 2
  v = fmaxf(v, SWZ_XOR(v, 0x101f));  // xor 4
  v = fmaxf(v, SWZ_XOR(v, 0x201f));  // xor 8
  return v;
}

__device__ __forceinline__ float red16_sum(float v) {
  v += SWZ_XOR(v, 0x041f);
  v += SWZ_XOR(v, 0x081f);
  v += SWZ_XOR(v, 0x101f);
  v += SWZ_XOR(v, 0x201f);
  return v;
}

__global__ __launch_bounds__(128) void head_attn_kernel(
    const float* __restrict__ x,  const float* __restrict__ Wq,
    const float* __restrict__ Wk, const float* __restrict__ Wv,
    float* __restrict__ out)
{
  // LDS staging (f16): q,k row-major [64][LDP]; v transposed [H][LDP over T];
  // ps = softmax probabilities row-major [64][LDP].
  __shared__ _Float16 qs [CTX * LDP];
  __shared__ _Float16 ks [CTX * LDP];
  __shared__ _Float16 vsT[HSZ * LDP];
  __shared__ _Float16 ps [CTX * LDP];

  const int b   = blockIdx.x;
  const int tid = threadIdx.x;
  const int w   = tid >> 5;       // wave id == row-tile index (0..3)
  const int lid = tid & 31;
  const int g   = lid >> 4;       // lane group (0: lanes 0-15, 1: lanes 16-31)
  const int ln  = lid & 15;       // N index (B/C/D layouts) / row index (A layout)

  // ---------------- A fragments of x (this wave's 16 rows), f32 -> f16 ------
  // A 16x32 f16 layout: lane (g,ln) row = ln, halves i -> K = 32kc+16g+i
  // (one contiguous 16-float run).
  const float* xr = x + ((size_t)b * CTX + 16 * w + ln) * EMB;
  v16h Ax[4];
#pragma unroll
  for (int kc = 0; kc < 4; ++kc) {
    const int c0 = 32 * kc + 16 * g;
#pragma unroll
    for (int i = 0; i < 8; ++i) {
      Ax[kc][i]     = (_Float16)xr[c0 + i];
      Ax[kc][8 + i] = (_Float16)xr[c0 + 8 + i];
    }
  }

  // ---------------- projections: q,k,v = x @ W^T ----------------------------
  // B 32x16 f16 layout: lane (g,ln) col = ln, halves i -> K = 32kc+16g+i.
  // B[k][n] = W[n][k]  -> per-lane 16 contiguous floats of W row (16nt+ln).
#pragma unroll
  for (int p = 0; p < 3; ++p) {
    const float* W  = (p == 0) ? Wq : (p == 1) ? Wk : Wv;
    _Float16* dst   = (p == 0) ? qs : (p == 1) ? ks : vsT;
#pragma unroll
    for (int nt = 0; nt < 4; ++nt) {
      v8f acc = {};
#pragma unroll
      for (int kc = 0; kc < 4; ++kc) {
        const float* wp = W + (16 * nt + ln) * EMB + 32 * kc + 16 * g;
        v16h bf;
#pragma unroll
        for (int i = 0; i < 16; ++i) bf[i] = (_Float16)wp[i];
        acc = wmma_f16f32(Ax[kc], bf, acc);
      }
      // D layout: VGPR r holds element (M = r + 8g, N = ln) of the 16x16 tile.
#pragma unroll
      for (int r = 0; r < 8; ++r) {
        const _Float16 h = (_Float16)acc[r];
        if (p == 2)  // v stored transposed: vsT[headdim][time]
          dst[(16 * nt + ln) * LDP + 16 * w + r + 8 * g] = h;
        else
          dst[(16 * w + r + 8 * g) * LDP + 16 * nt + ln] = h;
      }
    }
  }
  __syncthreads();  // k, vsT are consumed cross-wave

  // ---------------- wei = (q k^T) * scale, causal ---------------------------
  // A frags of q: per-lane row (16w+ln), two contiguous 8-half runs.
  v16h Aq[2];
#pragma unroll
  for (int kc = 0; kc < 2; ++kc) {
    const _Float16* qp = qs + (16 * w + ln) * LDP + 32 * kc + 16 * g;
#pragma unroll
    for (int i = 0; i < 8; ++i) { Aq[kc][i] = qp[i]; Aq[kc][8 + i] = qp[8 + i]; }
  }

  float wv[4][8];
#pragma unroll
  for (int st = 0; st < 4; ++st) {
    if (st <= w) {  // wave-uniform branch: EXEC stays all-ones around WMMA
      v8f acc = {};
#pragma unroll
      for (int kc = 0; kc < 2; ++kc) {
        // B[k][s] = k[s][k]: per-lane 16 contiguous halves of ks row (16st+ln)
        const _Float16* kp = ks + (16 * st + ln) * LDP + 32 * kc + 16 * g;
        v16h bf;
#pragma unroll
        for (int i = 0; i < 16; ++i) bf[i] = kp[i];
        acc = wmma_f16f32(Aq[kc], bf, acc);
      }
#pragma unroll
      for (int r = 0; r < 8; ++r) wv[st][r] = acc[r] * 0.125f;  // 1/sqrt(64)
    } else {
#pragma unroll
      for (int r = 0; r < 8; ++r) wv[st][r] = 0.0f;
    }
  }

  // ---------------- causal softmax over each row ----------------------------
  const float NEG_INF = -__builtin_inff();
#pragma unroll
  for (int r = 0; r < 8; ++r) {
    const int M = r + 8 * g;          // local row within this wave's tile
    float vals[4];
    float m = NEG_INF;
#pragma unroll
    for (int st = 0; st < 4; ++st) {
      float vvv;
      if (st < w)        vvv = wv[st][r];
      else if (st == w)  vvv = (ln <= M) ? wv[st][r] : NEG_INF;
      else               vvv = NEG_INF;
      vals[st] = vvv;
      m = fmaxf(m, vvv);
    }
    m = red16_max(m);
    float s = 0.0f;
#pragma unroll
    for (int st = 0; st < 4; ++st) {
      const float e = (vals[st] > NEG_INF) ? __expf(vals[st] - m) : 0.0f;
      vals[st] = e;
      s += e;
    }
    s = red16_sum(s);
    const float inv = 1.0f / s;
#pragma unroll
    for (int st = 0; st < 4; ++st)
      ps[(16 * w + M) * LDP + 16 * st + ln] = (_Float16)(vals[st] * inv);
  }
  // ps rows 16w..16w+15 are produced and consumed by the same wave; per-wave
  // in-order LDS (DScnt) ordering makes this safe without a barrier.

  // ---------------- out = P @ v ---------------------------------------------
  v16h Ap[2];
#pragma unroll
  for (int kc = 0; kc < 2; ++kc) {
    const _Float16* pp = ps + (16 * w + ln) * LDP + 32 * kc + 16 * g;
#pragma unroll
    for (int i = 0; i < 8; ++i) { Ap[kc][i] = pp[i]; Ap[kc][8 + i] = pp[8 + i]; }
  }

#pragma unroll
  for (int nt = 0; nt < 4; ++nt) {
    v8f acc = {};
#pragma unroll
    for (int kc = 0; kc < 2; ++kc) {
      // B[s][h] = v[s][h] = vsT[h][s]: contiguous halves along vsT row (16nt+ln)
      const _Float16* vp = vsT + (16 * nt + ln) * LDP + 32 * kc + 16 * g;
      v16h bf;
#pragma unroll
      for (int i = 0; i < 16; ++i) bf[i] = vp[i];
      acc = wmma_f16f32(Ap[kc], bf, acc);
    }
#pragma unroll
    for (int r = 0; r < 8; ++r)
      out[((size_t)b * CTX + 16 * w + r + 8 * g) * HSZ + 16 * nt + ln] = acc[r];
  }
}

extern "C" void kernel_launch(void* const* d_in, const int* in_sizes, int n_in,
                              void* d_out, int out_size, void* d_ws, size_t ws_size,
                              hipStream_t stream) {
  const float* x  = (const float*)d_in[0];
  const float* Wq = (const float*)d_in[1];
  const float* Wk = (const float*)d_in[2];
  const float* Wv = (const float*)d_in[3];
  float* out = (float*)d_out;
  const int B = in_sizes[0] / (CTX * EMB);   // 4096
  head_attn_kernel<<<dim3(B), dim3(128), 0, stream>>>(x, Wq, Wk, Wv, out);
  (void)n_in; (void)out_size; (void)d_ws; (void)ws_size;
}
